// Encoder_Core_78563541778981
// MI455X (gfx1250) — compile-verified, matrix operand order checked by reading, not run on hardware
//
#include <hip/hip_runtime.h>
#include <hip/hip_bf16.h>

typedef __attribute__((ext_vector_type(16))) _Float16 v16h;
typedef __attribute__((ext_vector_type(8)))  float    v8f;

#define NN   100000
#define EE   1200000
#define DD   64
#define LL   5
#define GG   128
#define LD   320   // L*D

// ---------------- helpers ----------------

__device__ __forceinline__ void atomicAddF(float* p, float v) {
  unsafeAtomicAdd(p, v);   // hardware global_atomic_add_f32
}

// Load 16 halves of the A matrix (16x32 f16 tile, ISA 7.12.2 layout) from an
// fp32 row. p points at A_row + kbase + k0 where k0 = (lane>>4)*8.
// halves 0..7  <- K = k0+0..7      (contiguous 8 floats)
// halves 8..15 <- K = k0+16+0..7   (contiguous 8 floats at +16)
__device__ __forceinline__ v16h load_a_f32(const float* __restrict__ p) {
  float4 f0 = *(const float4*)(p);
  float4 f1 = *(const float4*)(p + 4);
  float4 f2 = *(const float4*)(p + 16);
  float4 f3 = *(const float4*)(p + 20);
  v16h a;
  a[0]=(_Float16)f0.x;  a[1]=(_Float16)f0.y;  a[2]=(_Float16)f0.z;  a[3]=(_Float16)f0.w;
  a[4]=(_Float16)f1.x;  a[5]=(_Float16)f1.y;  a[6]=(_Float16)f1.z;  a[7]=(_Float16)f1.w;
  a[8]=(_Float16)f2.x;  a[9]=(_Float16)f2.y;  a[10]=(_Float16)f2.z; a[11]=(_Float16)f2.w;
  a[12]=(_Float16)f3.x; a[13]=(_Float16)f3.y; a[14]=(_Float16)f3.z; a[15]=(_Float16)f3.w;
  return a;
}

// ---------------- kernels ----------------

// Convert fp32 weight W[K x N] (row-major) into WMMA B-fragment order:
// pack[((ntile*(K/32) + kc)*32 + lane)*16 + j] = (f16) B[k][n]
//   n  = ntile*16 + (lane & 15)
//   k0 = (lane>>4)*8
//   k  = kc*32 + k0 + (j<8 ? j : 8+j)      (j>=8 -> k0+16+(j-8))
// Each lane's 16 halves are then contiguous (32B): one vector load per fragment.
__global__ void pack_b_kernel(const float* __restrict__ W, _Float16* __restrict__ pack,
                              int K, int N) {
  int gid = blockIdx.x * blockDim.x + threadIdx.x;
  if (gid >= K * N) return;
  int kchunks = K >> 5;
  int j     = gid & 15;
  int lane  = (gid >> 4) & 31;
  int chunk = gid >> 9;                 // ntile*kchunks + kc
  int kc    = chunk % kchunks;
  int ntile = chunk / kchunks;
  int n  = ntile * 16 + (lane & 15);
  int k0 = (lane >> 4) << 3;
  int k  = kc * 32 + k0 + ((j < 8) ? j : (8 + j));
  pack[gid] = (_Float16)W[(size_t)k * N + n];
}

__global__ void zero_f32_kernel(float* __restrict__ p, int n) {
  int i = blockIdx.x * blockDim.x + threadIdx.x;
  if (i < n) p[i] = 0.0f;
}

__global__ void copy_f32x4_kernel(const float* __restrict__ src, float* __restrict__ dst, int n4) {
  int i = blockIdx.x * blockDim.x + threadIdx.x;
  if (i < n4) ((float4*)dst)[i] = ((const float4*)src)[i];
}

// out[M x 64] = opt_relu(A[M x 64] @ W + bias), W pre-packed in B-fragment order.
// One wave per 16-row tile; A preloaded (2 x v16h), loop over 4 column tiles.
__global__ void wmma_gemm64_kernel(const float* __restrict__ A,
                                   const _Float16* __restrict__ Wpack,
                                   const float* __restrict__ bias,
                                   float* __restrict__ out,
                                   int mtiles, int relu) {
  int wid = blockIdx.x * (blockDim.x >> 5) + (threadIdx.x >> 5);
  if (wid >= mtiles) return;                 // wave-uniform exit (EXEC stays all-1s)
  int lane = threadIdx.x & 31;
  int m    = lane & 15;
  int hi   = lane >> 4;
  int k0   = hi << 3;                        // 0 or 8

  const float* arow = A + (size_t)(wid * 16 + m) * DD;
  v16h a0 = load_a_f32(arow + 0  + k0);      // K = 0..31
  v16h a1 = load_a_f32(arow + 32 + k0);      // K = 32..63

  const v16h* Bp = (const v16h*)Wpack;       // [ntile][kc][lane] fragments, 32B each

#pragma unroll
  for (int nt = 0; nt < 4; ++nt) {
    v16h b0 = Bp[(nt * 2 + 0) * 32 + lane];
    v16h b1 = Bp[(nt * 2 + 1) * 32 + lane];
    v8f acc = {};
    acc = __builtin_amdgcn_wmma_f32_16x16x32_f16(false, a0, false, b0, (short)0, acc, false, false);
    acc = __builtin_amdgcn_wmma_f32_16x16x32_f16(false, a1, false, b1, (short)0, acc, false, false);
    int ncol = nt * 16 + (lane & 15);
    float bv = bias[ncol];
    float* orow = out + (size_t)(wid * 16 + hi * 8) * DD + ncol;
#pragma unroll
    for (int r = 0; r < 8; ++r) {
      float v = acc[r] + bv;
      if (relu) v = fmaxf(v, 0.0f);
      orow[(size_t)r * DD] = v;
    }
  }
}

// Generic WMMA GEMM: out[M x N] = opt_relu(A[M x K] @ W + bias), W pre-packed.
// K % 32 == 0, M % 16 == 0, N % 16 == 0. One wave per 16x16 output tile.
__global__ void wmma_gemm_generic_kernel(const float* __restrict__ A,
                                         const _Float16* __restrict__ Wpack,
                                         const float* __restrict__ bias,
                                         float* __restrict__ out,
                                         int M, int K, int N, int relu) {
  int ntilesN = N >> 4;
  int total   = (M >> 4) * ntilesN;
  int wid = blockIdx.x * (blockDim.x >> 5) + (threadIdx.x >> 5);
  if (wid >= total) return;                  // wave-uniform
  int mt = wid / ntilesN, nt = wid - mt * ntilesN;
  int lane = threadIdx.x & 31;
  int hi   = lane >> 4;
  int k0   = hi << 3;
  int ncol = nt * 16 + (lane & 15);
  const float* arow = A + (size_t)(mt * 16 + (lane & 15)) * K;
  const v16h* Bp = (const v16h*)Wpack;
  int kchunks = K >> 5;

  v8f acc = {};
  for (int kc = 0; kc < kchunks; ++kc) {
    v16h a = load_a_f32(arow + kc * 32 + k0);
    v16h b = Bp[(nt * kchunks + kc) * 32 + lane];
    acc = __builtin_amdgcn_wmma_f32_16x16x32_f16(false, a, false, b, (short)0, acc, false, false);
  }
  float bv = bias[ncol];
  float* orow = out + (size_t)(mt * 16 + hi * 8) * N + ncol;
#pragma unroll
  for (int r = 0; r < 8; ++r) {
    float v = acc[r] + bv;
    if (relu) v = fmaxf(v, 0.0f);
    orow[(size_t)r * N] = v;
  }
}

// z[dst[e]] += h[src[e]] ; 16 threads per edge, float4 each (64 feats)
__global__ void scatter_add_kernel(const float* __restrict__ h,
                                   const int* __restrict__ src,
                                   const int* __restrict__ dst,
                                   float* __restrict__ z, int ne) {
  int gid = blockIdx.x * blockDim.x + threadIdx.x;
  int e = gid >> 4;
  if (e >= ne) return;
  int part = (gid & 15) * 4;
  int s = src[e], d = dst[e];
  float4 v = *(const float4*)(h + (size_t)s * DD + part);
  float* zp = z + (size_t)d * DD + part;
  atomicAddF(zp + 0, v.x);
  atomicAddF(zp + 1, v.y);
  atomicAddF(zp + 2, v.z);
  atomicAddF(zp + 3, v.w);
}

// per-column sum and sum-of-squares over all rows -> stats[0:64]=sum, stats[64:128]=sumsq
__global__ void bn_stats_kernel(const float* __restrict__ u, float* __restrict__ stats, int nrows) {
  __shared__ float ss[256];
  __shared__ float sq[256];
  int d = threadIdx.x & 63;
  int rlane = threadIdx.x >> 6;              // 0..3
  float s = 0.0f, q = 0.0f;
  for (int r = blockIdx.x * 4 + rlane; r < nrows; r += gridDim.x * 4) {
    float v = u[(size_t)r * DD + d];
    s += v; q += v * v;
  }
  ss[threadIdx.x] = s; sq[threadIdx.x] = q;
  __syncthreads();
  if (threadIdx.x < 64) {
    s = ss[d] + ss[64 + d] + ss[128 + d] + ss[192 + d];
    q = sq[d] + sq[64 + d] + sq[128 + d] + sq[192 + d];
    atomicAddF(&stats[d], s);
    atomicAddF(&stats[64 + d], q);
  }
}

// h_out = gamma*(u-mean)*rsqrt(var+eps)+beta ; pooled[batch[n], layer*64+d] += h_out
__global__ void bn_apply_pool_kernel(const float* __restrict__ u,
                                     const float* __restrict__ stats,
                                     const float* __restrict__ gamma,
                                     const float* __restrict__ beta,
                                     const int* __restrict__ batch,
                                     float* __restrict__ hout,
                                     float* __restrict__ pooled,
                                     int layer, int nrows) {
  int gid = blockIdx.x * blockDim.x + threadIdx.x;
  if (gid >= nrows * DD) return;
  int d = gid & 63;
  int n = gid >> 6;
  float inv_n = 1.0f / (float)nrows;
  float mean = stats[d] * inv_n;
  float var  = stats[64 + d] * inv_n - mean * mean;
  float inv  = rsqrtf(var + 1e-5f);
  float v = gamma[d] * (u[gid] - mean) * inv + beta[d];
  hout[gid] = v;
  atomicAddF(&pooled[(size_t)batch[n] * LD + layer * DD + d], v);
}

// row-wise L2 normalize: out[row] = in[row] / max(||in[row]||, 1e-12)
__global__ void l2norm_kernel(const float* __restrict__ in, float* __restrict__ out, int cols) {
  __shared__ float red[256];
  int row = blockIdx.x;
  float s = 0.0f;
  for (int c = threadIdx.x; c < cols; c += blockDim.x) {
    float v = in[(size_t)row * cols + c];
    s += v * v;
  }
  red[threadIdx.x] = s;
  __syncthreads();
  for (int off = 128; off > 0; off >>= 1) {
    if ((int)threadIdx.x < off) red[threadIdx.x] += red[threadIdx.x + off];
    __syncthreads();
  }
  float scale = 1.0f / fmaxf(sqrtf(red[0]), 1e-12f);
  for (int c = threadIdx.x; c < cols; c += blockDim.x)
    out[(size_t)row * cols + c] = in[(size_t)row * cols + c] * scale;
}

// ---------------- host launch ----------------

extern "C" void kernel_launch(void* const* d_in, const int* in_sizes, int n_in,
                              void* d_out, int out_size, void* d_ws, size_t ws_size,
                              hipStream_t stream) {
  const float* x     = (const float*)d_in[0];
  const int*   ei    = (const int*)d_in[1];     // [2, E]
  const int*   batch = (const int*)d_in[2];
  const float* Ws    = (const float*)d_in[3];
  const float* bs    = (const float*)d_in[4];
  const float* W1    = (const float*)d_in[5];   // [L,64,64]
  const float* b1    = (const float*)d_in[6];
  const float* W2    = (const float*)d_in[7];
  const float* b2    = (const float*)d_in[8];
  const float* gamma = (const float*)d_in[9];
  const float* beta  = (const float*)d_in[10];
  const float* Wp1   = (const float*)d_in[11];  // [320,320]
  const float* bp1   = (const float*)d_in[12];
  const float* Wp2   = (const float*)d_in[13];
  const float* bp2   = (const float*)d_in[14];
  float* out = (float*)d_out;

  const int* src = ei;
  const int* dst = ei + EE;

  // workspace carve-up (~78 MB)
  char* ws = (char*)d_ws;
  size_t off = 0;
  auto carve = [&](size_t bytes) { size_t o = off; off = (off + bytes + 255) & ~(size_t)255; return o; };
  _Float16* Ws16  = (_Float16*)(ws + carve(4096 * 2));
  _Float16* W1_16 = (_Float16*)(ws + carve(LL * 4096 * 2));
  _Float16* W2_16 = (_Float16*)(ws + carve(LL * 4096 * 2));
  _Float16* Wp1_16= (_Float16*)(ws + carve((size_t)LD * LD * 2));
  _Float16* Wp2_16= (_Float16*)(ws + carve((size_t)LD * LD * 2));
  float* bufA   = (float*)(ws + carve((size_t)NN * DD * 4));  // h
  float* bufB   = (float*)(ws + carve((size_t)NN * DD * 4));  // z / u
  float* bufC   = (float*)(ws + carve((size_t)NN * DD * 4));  // t
  float* pooled = (float*)(ws + carve((size_t)GG * LD * 4));
  float* stats  = (float*)(ws + carve(128 * 4));
  float* y1     = (float*)(ws + carve((size_t)GG * LD * 4));
  float* y      = (float*)(ws + carve((size_t)GG * LD * 4));

  // 1) weights -> f16 in WMMA B-fragment order
  pack_b_kernel<<<(4096 + 255) / 256, 256, 0, stream>>>(Ws, Ws16, DD, DD);
  for (int l = 0; l < LL; ++l) {
    pack_b_kernel<<<(4096 + 255) / 256, 256, 0, stream>>>(W1 + l * 4096, W1_16 + l * 4096, DD, DD);
    pack_b_kernel<<<(4096 + 255) / 256, 256, 0, stream>>>(W2 + l * 4096, W2_16 + l * 4096, DD, DD);
  }
  pack_b_kernel<<<(LD * LD + 255) / 256, 256, 0, stream>>>(Wp1, Wp1_16, LD, LD);
  pack_b_kernel<<<(LD * LD + 255) / 256, 256, 0, stream>>>(Wp2, Wp2_16, LD, LD);

  const int mtiles = NN / 16;                       // 6250
  const int gemm_blocks = (mtiles + 7) / 8;         // 8 waves / block

  // 2) encoder: h = x @ Ws + bs
  wmma_gemm64_kernel<<<gemm_blocks, 256, 0, stream>>>(x, Ws16, bs, bufA, mtiles, 0);

  // 3) pooled = 0
  zero_f32_kernel<<<(GG * LD + 255) / 256, 256, 0, stream>>>(pooled, GG * LD);

  // 4) GIN layers
  for (int l = 0; l < LL; ++l) {
    // z = h  (then scatter-add neighbors)
    copy_f32x4_kernel<<<(NN * DD / 4 + 255) / 256, 256, 0, stream>>>(bufA, bufB, NN * DD / 4);
    scatter_add_kernel<<<(EE * 16 + 255) / 256, 256, 0, stream>>>(bufA, src, dst, bufB, EE);
    // t = relu(z @ W1 + b1)
    wmma_gemm64_kernel<<<gemm_blocks, 256, 0, stream>>>(bufB, W1_16 + l * 4096, b1 + l * 64, bufC, mtiles, 1);
    // u = relu(t @ W2 + b2)   (reference: relu applied after conv output)
    wmma_gemm64_kernel<<<gemm_blocks, 256, 0, stream>>>(bufC, W2_16 + l * 4096, b2 + l * 64, bufB, mtiles, 1);
    // batchnorm (training-mode stats) + per-graph add-pool
    zero_f32_kernel<<<1, 128, 0, stream>>>(stats, 128);
    bn_stats_kernel<<<512, 256, 0, stream>>>(bufB, stats, NN);
    bn_apply_pool_kernel<<<(NN * DD + 255) / 256, 256, 0, stream>>>(
        bufB, stats, gamma + l * 64, beta + l * 64, batch, bufA, pooled, l, NN);
  }

  // 5) projection head: y = relu(pooled @ Wp1 + bp1) @ Wp2 + bp2
  {
    int total = (GG / 16) * (LD / 16);              // 160 wave-tiles
    int blocks = (total + 7) / 8;
    wmma_gemm_generic_kernel<<<blocks, 256, 0, stream>>>(pooled, Wp1_16, bp1, y1, GG, LD, LD, 1);
    wmma_gemm_generic_kernel<<<blocks, 256, 0, stream>>>(y1, Wp2_16, bp2, y, GG, LD, LD, 0);
  }

  // 6) outputs: [l2norm(y), l2norm(pooled)]
  l2norm_kernel<<<GG, 256, 0, stream>>>(y, out, LD);
  l2norm_kernel<<<GG, 256, 0, stream>>>(pooled, out + GG * LD, LD);
}
